// ImageAugmentation_33045478375536
// MI455X (gfx1250) — compile-verified, matrix operand order checked by reading, not run on hardware
//
#include <hip/hip_runtime.h>

#define IMG_S 224
#define NPIX (IMG_S * IMG_S)   // 50176

typedef float    v2f  __attribute__((ext_vector_type(2)));
typedef float    v8f  __attribute__((ext_vector_type(8)));
typedef _Float16 v16h __attribute__((ext_vector_type(16)));

__device__ __forceinline__ float clip01(float v) { return fminf(fmaxf(v, 0.0f), 1.0f); }

// ---------------------------------------------------------------------------
// Kernel 1: per-image mean of grayscale(clip(bf*img)). One block per image.
// Per-lane float2 partials are collapsed per-wave by a single f32 WMMA
// (B = all-ones -> D rows are row-sums of A; layout-independent result).
// ---------------------------------------------------------------------------
__global__ __launch_bounds__(256) void mean_gray_kernel(
    const float* __restrict__ img, const float* __restrict__ u_bright,
    float* __restrict__ mean_out)
{
  __shared__ float smem[16];
  const int b   = blockIdx.x;
  const int tid = threadIdx.x;
  const float bf = 1.0f + (2.0f * u_bright[b] - 1.0f) * 0.2f;

  const float* pr = img + (size_t)b * (3 * NPIX);
  const float* pg = pr + NPIX;
  const float* pb = pr + 2 * NPIX;

  float ax = 0.0f, ay = 0.0f;
  for (int it = 0; it < NPIX / 512; ++it) {   // 98 iterations, 2 pixels/thread
    int p = it * 512 + tid * 2;
    float2 rv = *(const float2*)(pr + p);
    float2 gv = *(const float2*)(pg + p);
    float2 bv = *(const float2*)(pb + p);
    ax += 0.299f * clip01(rv.x * bf) + 0.587f * clip01(gv.x * bf) + 0.114f * clip01(bv.x * bf);
    ay += 0.299f * clip01(rv.y * bf) + 0.587f * clip01(gv.y * bf) + 0.114f * clip01(bv.y * bf);
  }

  v8f c = {};
  v8f d;
  float post = 1.0f;
#if __has_builtin(__builtin_amdgcn_wmma_f32_16x16x4_f32)
  {
    v2f a;    a.x = ax;     a.y = ay;
    v2f ones; ones.x = 1.0f; ones.y = 1.0f;
    d = __builtin_amdgcn_wmma_f32_16x16x4_f32(false, a, false, ones, (short)0, c, false, false);
  }
#else
  {
    // Fallback: codegen-confirmed f16 WMMA; pre-scale partials to keep f16 precision.
    const float SC = 1.0f / 1024.0f;
    v16h ah, bh;
    #pragma unroll
    for (int i = 0; i < 16; ++i) { ah[i] = (_Float16)0.0f; bh[i] = (_Float16)1.0f; }
    ah[0] = (_Float16)(ax * SC);
    ah[1] = (_Float16)(ay * SC);
    d = __builtin_amdgcn_wmma_f32_16x16x32_f16(false, ah, false, bh, (short)0, c, false, false);
    post = 1024.0f;
  }
#endif
  // D layout: VGPR r, lanes 0-15 -> M=r; lanes 16-31 -> M=8+r. Sum of the 8
  // D VGPRs gives rows 0-7 total in lanes 0-15 and rows 8-15 total in lanes 16-31.
  float s = (d[0] + d[1] + d[2] + d[3] + d[4] + d[5] + d[6] + d[7]) * post;

  const int lane = tid & 31, wave = tid >> 5;
  if ((lane & 15) == 0) smem[wave * 2 + (lane >> 4)] = s;
  __syncthreads();
  if (tid == 0) {
    float tot = 0.0f;
    #pragma unroll
    for (int i = 0; i < 16; ++i) tot += smem[i];
    mean_out[b] = tot / (float)NPIX;
  }
}

// ---------------------------------------------------------------------------
// Color jitter (brightness -> contrast -> saturation -> hue), matching the
// reference op-for-op including exact-equality hue branch precedence.
// ---------------------------------------------------------------------------
struct JParams { float bf, cf, sf, hf, mg; };

__device__ __forceinline__ void color_jitter(float& r, float& g, float& b, const JParams& P)
{
  r = clip01(r * P.bf); g = clip01(g * P.bf); b = clip01(b * P.bf);

  float omc = 1.0f - P.cf;
  r = clip01(P.cf * r + omc * P.mg);
  g = clip01(P.cf * g + omc * P.mg);
  b = clip01(P.cf * b + omc * P.mg);

  float gray = 0.299f * r + 0.587f * g + 0.114f * b;
  float oms = 1.0f - P.sf;
  r = clip01(P.sf * r + oms * gray);
  g = clip01(P.sf * g + oms * gray);
  b = clip01(P.sf * b + oms * gray);

  float maxc = fmaxf(r, fmaxf(g, b));
  float minc = fminf(r, fminf(g, b));
  float v  = maxc;
  float cr = maxc - minc;
  bool  eqc = (cr == 0.0f);
  float s   = cr / (eqc ? 1.0f : maxc);
  float crd = eqc ? 1.0f : cr;
  float rc = (maxc - r) / crd;
  float gc = (maxc - g) / crd;
  float bc = (maxc - b) / crd;

  float h;
  if (maxc == r)      h = bc - gc;
  else if (maxc == g) h = 2.0f + rc - bc;
  else                h = 4.0f + gc - rc;

  h = h / 6.0f + 1.0f;
  h = h - floorf(h);
  h = h + P.hf;
  h = h - floorf(h);

  float h6 = h * 6.0f;
  float fi = floorf(h6);
  float f  = h6 - fi;
  int   i  = ((int)fi) % 6;
  float p = v * (1.0f - s);
  float q = v * (1.0f - s * f);
  float t = v * (1.0f - s * (1.0f - f));

  float r2, g2, b2;
  switch (i) {
    case 0:  r2 = v; g2 = t; b2 = p; break;
    case 1:  r2 = q; g2 = v; b2 = p; break;
    case 2:  r2 = p; g2 = v; b2 = t; break;
    case 3:  r2 = p; g2 = q; b2 = v; break;
    case 4:  r2 = t; g2 = p; b2 = v; break;
    default: r2 = v; g2 = p; b2 = q; break;
  }
  r = clip01(r2); g = clip01(g2); b = clip01(b2);
}

// ---------------------------------------------------------------------------
// Kernel 2: fused jitter + crop-resize. One block = 256 contiguous spatial
// positions of a single image (uniform per-image branches, no divergence).
// ---------------------------------------------------------------------------
__global__ __launch_bounds__(256) void augment_kernel(
    const float* __restrict__ img,
    const float* __restrict__ u_jitter, const float* __restrict__ u_bright,
    const float* __restrict__ u_contrast, const float* __restrict__ u_sat,
    const float* __restrict__ u_hue, const float* __restrict__ u_crop,
    const float* __restrict__ u_scale, const float* __restrict__ u_top,
    const float* __restrict__ u_left, const float* __restrict__ mean_gray,
    float* __restrict__ out)
{
  const int bpi = NPIX / 256;                      // 196 blocks per image
  const int b   = blockIdx.x / bpi;
  const int pix = (blockIdx.x - b * bpi) * 256 + threadIdx.x;
  const int y = pix / IMG_S;
  const int x = pix - y * IMG_S;

  JParams P;
  P.bf = 1.0f + (2.0f * u_bright[b]   - 1.0f) * 0.2f;
  P.cf = 1.0f + (2.0f * u_contrast[b] - 1.0f) * 0.2f;
  P.sf = 1.0f + (2.0f * u_sat[b]      - 1.0f) * 0.2f;
  P.hf =        (2.0f * u_hue[b]      - 1.0f) * 0.05f;
  P.mg = mean_gray[b];
  const bool jitter_on = u_jitter[b] < 0.8f;
  const bool crop_on   = u_crop[b]   < 0.5f;

  const float* base = img + (size_t)b * (3 * NPIX);
  float r, g, bl;

  if (crop_on) {
    float scale = 0.85f + 0.15f * u_scale[b];
    float ch = floorf(224.0f * scale);
    float cw = ch;  // H == W, same formula
    float top  = fminf(fmaxf(floorf(u_top[b]  * (224.0f - ch + 1.0f)), 0.0f), 224.0f - ch);
    float left = fminf(fmaxf(floorf(u_left[b] * (224.0f - cw + 1.0f)), 0.0f), 224.0f - cw);

    float ys = fminf(fmaxf(top  + ((float)y + 0.5f) * ch / 224.0f - 0.5f, 0.0f), 223.0f);
    float xs = fminf(fmaxf(left + ((float)x + 0.5f) * cw / 224.0f - 0.5f, 0.0f), 223.0f);

    float y0f = floorf(ys), x0f = floorf(xs);
    float wy = ys - y0f, wx = xs - x0f;
    int y0 = (int)y0f, x0 = (int)x0f;
    int y1 = min(y0 + 1, IMG_S - 1);
    int x1 = min(x0 + 1, IMG_S - 1);

    int off[4] = { y0 * IMG_S + x0, y0 * IMG_S + x1,
                   y1 * IMG_S + x0, y1 * IMG_S + x1 };
    float rr[4], gg[4], bb[4];
    #pragma unroll
    for (int k = 0; k < 4; ++k) {
      rr[k] = base[off[k]];
      gg[k] = base[NPIX + off[k]];
      bb[k] = base[2 * NPIX + off[k]];
    }
    if (jitter_on) {
      #pragma unroll
      for (int k = 0; k < 4; ++k) color_jitter(rr[k], gg[k], bb[k], P);
    }
    float omx = 1.0f - wx, omy = 1.0f - wy;
    r  = omy * (omx * rr[0] + wx * rr[1]) + wy * (omx * rr[2] + wx * rr[3]);
    g  = omy * (omx * gg[0] + wx * gg[1]) + wy * (omx * gg[2] + wx * gg[3]);
    bl = omy * (omx * bb[0] + wx * bb[1]) + wy * (omx * bb[2] + wx * bb[3]);
  } else {
    r  = base[pix];
    g  = base[NPIX + pix];
    bl = base[2 * NPIX + pix];
    if (jitter_on) color_jitter(r, g, bl, P);
  }

  float* ob = out + (size_t)b * (3 * NPIX);
  ob[pix]             = r;
  ob[NPIX + pix]      = g;
  ob[2 * NPIX + pix]  = bl;
}

// ---------------------------------------------------------------------------
extern "C" void kernel_launch(void* const* d_in, const int* in_sizes, int n_in,
                              void* d_out, int out_size, void* d_ws, size_t ws_size,
                              hipStream_t stream)
{
  const float* img        = (const float*)d_in[0];
  const float* u_jitter   = (const float*)d_in[1];
  const float* u_bright   = (const float*)d_in[2];
  const float* u_contrast = (const float*)d_in[3];
  const float* u_sat      = (const float*)d_in[4];
  const float* u_hue      = (const float*)d_in[5];
  const float* u_crop     = (const float*)d_in[6];
  const float* u_scale    = (const float*)d_in[7];
  const float* u_top      = (const float*)d_in[8];
  const float* u_left     = (const float*)d_in[9];
  float* out = (float*)d_out;
  float* mg  = (float*)d_ws;   // B floats of scratch

  const int B = in_sizes[1];

  mean_gray_kernel<<<B, 256, 0, stream>>>(img, u_bright, mg);

  const int bpi = NPIX / 256;  // 196
  augment_kernel<<<B * bpi, 256, 0, stream>>>(
      img, u_jitter, u_bright, u_contrast, u_sat, u_hue, u_crop,
      u_scale, u_top, u_left, mg, out);
}
